// SAConvLSTM_62534723830305
// MI455X (gfx1250) — matrix operand order repaired and checked by reading, not compiled, not run on hardware
//
#include <hip/hip_runtime.h>
#include <hip/hip_bf16.h>
#include <math.h>

typedef _Float16 half_t;
typedef __attribute__((ext_vector_type(8)))  _Float16 v8h;
typedef __attribute__((ext_vector_type(16))) _Float16 v16h;
typedef __attribute__((ext_vector_type(8)))  float    v8f;

#define BATCH 4
#define SEQ   16
#define NPIX  1024      // H*W = 32*32
#define NCOL  4096      // BATCH*NPIX

#if defined(__gfx1250__)
#define CDNA5_ASYNC 1
#else
#define CDNA5_ASYNC 0
#endif

// ---------------------------------------------------------------------------
// WMMA tile helpers (wave32, 16x16x32 f16 -> f32)
// ---------------------------------------------------------------------------
__device__ inline v16h load_a_tile(const half_t* __restrict__ p, int ld, int lane) {
    // A 16x32 f16 row-major tile. lane<16: K {0..7}+{16..23}; lane>=16: K {8..15}+{24..31}
    int r  = lane & 15;
    int hi = lane >> 4;
    const half_t* q = p + (size_t)r * ld + hi * 8;
    union { v16h v; v8h h[2]; } u;
    u.h[0] = *(const v8h*)(q);
    u.h[1] = *(const v8h*)(q + 16);
    return u.v;
}

__device__ inline v16h load_b_tile(const half_t* __restrict__ p, int ld, int lane) {
    // B stored transposed (N x K row-major). lane<16: K {0..7}+{16..23}; lane>=16: K {8..15}+{24..31}
    int c  = lane & 15;
    int hi = lane >> 4;
    const half_t* q = p + (size_t)c * ld + hi * 8;
    union { v16h v; v8h h[2]; } u;
    u.h[0] = *(const v8h*)(q);
    u.h[1] = *(const v8h*)(q + 16);
    return u.v;
}

__device__ inline v8f wmma16(v16h a, v16h b, v8f c) {
    return __builtin_amdgcn_wmma_f32_16x16x32_f16(false, a, false, b, (short)0, c, false, false);
}

__device__ inline float sigm(float x) { return 1.f / (1.f + expf(-x)); }

#if CDNA5_ASYNC
// Async bulk copy global -> LDS (16 bytes per lane per issue), tracked by ASYNCcnt.
__device__ inline void async_load_b128(unsigned int lds_addr, unsigned long long gaddr) {
    asm volatile("global_load_async_to_lds_b128 %0, %1, off"
                 :: "v"(lds_addr), "v"(gaddr) : "memory");
}
__device__ inline void wait_asynccnt0() {
    asm volatile("s_wait_asynccnt 0" ::: "memory");
}
#endif

// ---------------------------------------------------------------------------
// Packing / init kernels
// ---------------------------------------------------------------------------
__global__ void k_f32_to_f16(const float* __restrict__ s, half_t* __restrict__ d, int n) {
    int i = blockIdx.x * blockDim.x + threadIdx.x;
    if (i < n) d[i] = (half_t)s[i];
}
__global__ void k_zero_h(half_t* d, int n) {
    int i = blockIdx.x * blockDim.x + threadIdx.x;
    if (i < n) d[i] = (half_t)0.f;
}
__global__ void k_zero_f(float* d, int n) {
    int i = blockIdx.x * blockDim.x + threadIdx.x;
    if (i < n) d[i] = 0.f;
}

// conv_w (256,128,3,3) f32 -> blocks[(mtile*9+tap)*4+kc][r*32+kk] f16 (A-tile layout)
__global__ void k_pack_conv(const float* __restrict__ w, half_t* __restrict__ dst) {
    int idx = blockIdx.x * blockDim.x + threadIdx.x;
    if (idx >= 256 * 128 * 9) return;
    int o   = idx / (128 * 9);
    int r2  = idx % (128 * 9);
    int ci  = r2 / 9;
    int tap = r2 % 9;
    int mtile = o >> 4, mr = o & 15;
    int kc = ci >> 5,  kk = ci & 31;
    dst[(size_t)(((mtile * 9 + tap) * 4 + kc)) * 512 + mr * 32 + kk] = (half_t)w[idx];
}

// generic MxK f32 row-major -> stacked A-tile blocks [(mtile+moff)*KC+kc][r*32+k]
__global__ void k_pack_a(const float* __restrict__ src, half_t* __restrict__ dst,
                         int Msrc, int K, int moff) {
    int idx = blockIdx.x * blockDim.x + threadIdx.x;
    if (idx >= Msrc * K) return;
    int m = idx / K, k = idx % K;
    int KC = K >> 5;
    dst[(size_t)(((m >> 4) + moff) * KC + (k >> 5)) * 512 + (m & 15) * 32 + (k & 31)] =
        (half_t)src[idx];
}

// ---------------------------------------------------------------------------
// Conv gates: 9-tap implicit GEMM, M=256, N=4096, K=128 per tap (f16 WMMA)
// block = 128 threads (4 waves); one 16-pixel column tile per block
// ---------------------------------------------------------------------------
__global__ __launch_bounds__(128)
void k_conv(const half_t* __restrict__ Xf16, const half_t* __restrict__ Hf16,
            const half_t* __restrict__ WcP, const float* __restrict__ conv_b,
            float* __restrict__ gbuf, int t) {
    __shared__ __align__(16) half_t BT[16 * 32];
    const int tid  = threadIdx.x;
    const int lane = tid & 31;
    const int wave = tid >> 5;
    const int col0 = blockIdx.x * 16;
    const int b  = col0 >> 10;
    const int p0 = col0 & 1023;
    const int y  = p0 >> 5;
    const int x0 = p0 & 31;

    v8f acc[4];
    for (int j = 0; j < 4; ++j) acc[j] = {};

    const int n_st = tid & 15;   // pixel within tile (staging role)
    const int kq   = tid >> 4;   // 0..7

    for (int tap = 0; tap < 9; ++tap) {
        const int dy = tap / 3 - 1;
        const int dx = tap % 3 - 1;
        const int py = y + dy;
        const int px = x0 + n_st + dx;
        const bool valid = ((unsigned)py < 32u) && ((unsigned)px < 32u);
        const int pp = py * 32 + px;
        for (int kc = 0; kc < 4; ++kc) {
            __syncthreads();
            for (int j = 0; j < 4; ++j) {
                int k = kq * 4 + j;
                int c = kc * 32 + k;
                half_t v = (half_t)0.f;
                if (valid) {
                    v = (c < 64) ? Xf16[((size_t)(b * 64 + c) * SEQ + t) * NPIX + pp]
                                 : Hf16[(size_t)(b * 64 + (c - 64)) * NPIX + pp];
                }
                BT[n_st * 32 + k] = v;
            }
            __syncthreads();
            v16h bm = load_b_tile(&BT[0], 32, lane);
            for (int j = 0; j < 4; ++j) {
                int mt = wave * 4 + j;
                v16h a = load_a_tile(WcP + (size_t)((mt * 9 + tap) * 4 + kc) * 512, 32, lane);
                acc[j] = wmma16(a, bm, acc[j]);
            }
        }
    }
    const int colL = lane & 15;
    const int hi   = lane >> 4;
    for (int j = 0; j < 4; ++j)
        for (int r = 0; r < 8; ++r) {
            int m = (wave * 4 + j) * 16 + r + hi * 8;
            gbuf[(size_t)m * NCOL + col0 + colL] = acc[j][r] + conv_b[m];
        }
}

// ---------------------------------------------------------------------------
// LSTM elementwise gates
// ---------------------------------------------------------------------------
__global__ void k_lstm(const float* __restrict__ g, const float* __restrict__ cprev,
                       const float* __restrict__ Wci, const float* __restrict__ Wcf,
                       const float* __restrict__ Wco,
                       float* __restrict__ cnew, half_t* __restrict__ hT,
                       half_t* __restrict__ mT, half_t* __restrict__ catT2) {
    int idx = blockIdx.x * blockDim.x + threadIdx.x;
    if (idx >= BATCH * 64 * NPIX) return;
    int p  = idx & 1023;
    int ch = (idx >> 10) & 63;
    int b  = idx >> 16;
    int col = b * NPIX + p;
    float gi = g[(size_t)(0 * 64 + ch) * NCOL + col];
    float gf = g[(size_t)(1 * 64 + ch) * NCOL + col];
    float gg = g[(size_t)(2 * 64 + ch) * NCOL + col];
    float go = g[(size_t)(3 * 64 + ch) * NCOL + col];
    float c0 = cprev[idx];
    int wp = ch * NPIX + p;
    float i  = sigm(gi + Wci[wp] * c0);
    float f  = sigm(gf + Wcf[wp] * c0);
    float cn = f * c0 + i * tanhf(gg);
    float o  = sigm(go + Wco[wp] * cn);
    float hn = o * tanhf(cn);
    cnew[idx] = cn;
    hT[(size_t)(b * NPIX + p) * 64 + ch]            = (half_t)hn;
    mT[(size_t)(b * NPIX + p) * 64 + ch]            = (half_t)cn;
    catT2[(size_t)(b * NPIX + p) * 192 + 128 + ch]  = (half_t)hn;
}

// ---------------------------------------------------------------------------
// Fused q/k/v projection from hT (B^T layout), M=96, K=64
// rows 0..15 -> qA (transposed, K-padded 32), 16..31 -> kT, 32..95 -> vh
// ---------------------------------------------------------------------------
__global__ __launch_bounds__(32)
void k_projH(const half_t* __restrict__ Wp, const half_t* __restrict__ hT,
             const float* __restrict__ qb, const float* __restrict__ kb,
             const float* __restrict__ vb,
             half_t* __restrict__ qA, half_t* __restrict__ kT, half_t* __restrict__ vh) {
    int lane = threadIdx.x;
    int mt = blockIdx.x, ctile = blockIdx.y, b = blockIdx.z;
    const half_t* Bbase = hT + (size_t)(b * NPIX + ctile * 16) * 64;
    v8f acc = {};
    for (int kc = 0; kc < 2; ++kc) {
        v16h a  = load_a_tile(Wp + (size_t)(mt * 2 + kc) * 512, 32, lane);
        v16h bm = load_b_tile(Bbase + kc * 32, 64, lane);
        acc = wmma16(a, bm, acc);
    }
    int col = ctile * 16 + (lane & 15);
    int hi  = lane >> 4;
    for (int r = 0; r < 8; ++r) {
        int row = mt * 16 + r + hi * 8;
        float v = acc[r];
        if (row < 16)
            qA[(size_t)(b * NPIX + col) * 32 + row]            = (half_t)(v + qb[row]);
        else if (row < 32)
            kT[(size_t)(b * NPIX + col) * 32 + (row - 16)]     = (half_t)(v + kb[row - 16]);
        else
            vh[(size_t)(b * 64 + (row - 32)) * NPIX + col]     = (half_t)(v + vb[row - 32]);
    }
}

// k2/v2 projection from mT, M=80: rows 0..15 -> kmT, 16..79 -> vm
__global__ __launch_bounds__(32)
void k_projM(const half_t* __restrict__ Wp, const half_t* __restrict__ mT,
             const float* __restrict__ kb, const float* __restrict__ vb,
             half_t* __restrict__ kmT, half_t* __restrict__ vm) {
    int lane = threadIdx.x;
    int mt = blockIdx.x, ctile = blockIdx.y, b = blockIdx.z;
    const half_t* Bbase = mT + (size_t)(b * NPIX + ctile * 16) * 64;
    v8f acc = {};
    for (int kc = 0; kc < 2; ++kc) {
        v16h a  = load_a_tile(Wp + (size_t)(mt * 2 + kc) * 512, 32, lane);
        v16h bm = load_b_tile(Bbase + kc * 32, 64, lane);
        acc = wmma16(a, bm, acc);
    }
    int col = ctile * 16 + (lane & 15);
    int hi  = lane >> 4;
    for (int r = 0; r < 8; ++r) {
        int row = mt * 16 + r + hi * 8;
        float v = acc[r];
        if (row < 16)
            kmT[(size_t)(b * NPIX + col) * 32 + row]       = (half_t)(v + kb[row]);
        else
            vm[(size_t)(b * 64 + (row - 16)) * NPIX + col] = (half_t)(v + vb[row - 16]);
    }
}

// ---------------------------------------------------------------------------
// Attention logits: (1024 x 1024) = qA (1024x32, K-padded) x kX^T, one WMMA/tile
// ---------------------------------------------------------------------------
__global__ __launch_bounds__(32)
void k_attn(const half_t* __restrict__ qA, const half_t* __restrict__ kX,
            float* __restrict__ logits) {
    int lane = threadIdx.x;
    int mt = blockIdx.x, ct = blockIdx.y, b = blockIdx.z;
    v16h a  = load_a_tile(qA + (size_t)(b * NPIX + mt * 16) * 32, 32, lane);
    v16h bm = load_b_tile(kX + (size_t)(b * NPIX + ct * 16) * 32, 32, lane);
    v8f acc = {};
    acc = wmma16(a, bm, acc);
    int col = ct * 16 + (lane & 15);
    int hi  = lane >> 4;
    for (int r = 0; r < 8; ++r) {
        int row = mt * 16 + r + hi * 8;
        logits[(size_t)(b * NPIX + row) * NPIX + col] = acc[r];
    }
}

// row softmax over 1024, f32 in -> f16 out
__global__ __launch_bounds__(256)
void k_softmax(const float* __restrict__ logits, half_t* __restrict__ attn) {
    __shared__ float red[256];
    int row = blockIdx.x;
    int tid = threadIdx.x;
    const float* src = logits + (size_t)row * NPIX;
    float v[4];
    float mx = -1e30f;
    for (int j = 0; j < 4; ++j) { v[j] = src[tid + j * 256]; mx = fmaxf(mx, v[j]); }
    red[tid] = mx; __syncthreads();
    for (int s = 128; s > 0; s >>= 1) { if (tid < s) red[tid] = fmaxf(red[tid], red[tid + s]); __syncthreads(); }
    mx = red[0]; __syncthreads();
    float sum = 0.f;
    for (int j = 0; j < 4; ++j) { v[j] = expf(v[j] - mx); sum += v[j]; }
    red[tid] = sum; __syncthreads();
    for (int s = 128; s > 0; s >>= 1) { if (tid < s) red[tid] += red[tid + s]; __syncthreads(); }
    float inv = 1.f / red[0];
    half_t* dst = attn + (size_t)row * NPIX;
    for (int j = 0; j < 4; ++j) dst[tid + j * 256] = (half_t)(v[j] * inv);
}

// ---------------------------------------------------------------------------
// z^T[n][c] = sum_m attn[n][m] * v[c][m]; K = 1024.
// Both operand strips (16 contiguous rows of attn, 16 contiguous rows of v;
// 32 KB each) are DMA'd into LDS with GLOBAL_LOAD_ASYNC_TO_LDS_B128 and the
// 32-step WMMA loop runs entirely out of LDS (ds_load_b128).
// ---------------------------------------------------------------------------
__global__ __launch_bounds__(32)
void k_zgemm(const half_t* __restrict__ attn, const half_t* __restrict__ vmat,
             half_t* __restrict__ zcatT, int zoff) {
    __shared__ __align__(16) half_t As[16 * 1024];
    __shared__ __align__(16) half_t Bs[16 * 1024];
    int lane = threadIdx.x;
    int mt = blockIdx.x, nt = blockIdx.y, b = blockIdx.z;
    const half_t* Ag = attn + (size_t)(b * NPIX + mt * 16) * NPIX;   // contiguous 32KB strip
    const half_t* Bg = vmat + (size_t)(b * 64 + nt * 16) * NPIX;     // contiguous 32KB strip

#if CDNA5_ASYNC
    unsigned int aL = (unsigned int)(size_t)(void*)As;
    unsigned int bL = (unsigned int)(size_t)(void*)Bs;
    for (int it = 0; it < 64; ++it) {
        int chunk = it * 32 + lane;                        // 2048 x 16B per strip
        unsigned long long ga = (unsigned long long)(size_t)((const char*)Ag + chunk * 16);
        unsigned long long gb = (unsigned long long)(size_t)((const char*)Bg + chunk * 16);
        async_load_b128(aL + chunk * 16, ga);
        async_load_b128(bL + chunk * 16, gb);
    }
    wait_asynccnt0();
#else
    for (int it = 0; it < 64; ++it) {
        int chunk = it * 32 + lane;
        *(v8h*)((char*)As + chunk * 16) = *(const v8h*)((const char*)Ag + chunk * 16);
        *(v8h*)((char*)Bs + chunk * 16) = *(const v8h*)((const char*)Bg + chunk * 16);
    }
#endif
    __syncthreads();

    v8f acc = {};
    for (int kc = 0; kc < 32; ++kc) {
        v16h a  = load_a_tile(As + kc * 32, NPIX, lane);
        v16h bm = load_b_tile(Bs + kc * 32, NPIX, lane);
        acc = wmma16(a, bm, acc);
    }
    int col = nt * 16 + (lane & 15);
    int hi  = lane >> 4;
    for (int r = 0; r < 8; ++r) {
        int row = mt * 16 + r + hi * 8;
        zcatT[(size_t)(b * NPIX + row) * 128 + zoff + col] = (half_t)acc[r];
    }
}

// zz = z_w(128x128) @ zcat; D written transposed into catT2 cols 0..127
__global__ __launch_bounds__(32)
void k_zz(const half_t* __restrict__ Wp, const half_t* __restrict__ zcatT,
          const float* __restrict__ zb, half_t* __restrict__ catT2) {
    int lane = threadIdx.x;
    int mt = blockIdx.x, ct = blockIdx.y, b = blockIdx.z;
    const half_t* Bbase = zcatT + (size_t)(b * NPIX + ct * 16) * 128;
    v8f acc = {};
    for (int kc = 0; kc < 4; ++kc) {
        v16h a  = load_a_tile(Wp + (size_t)(mt * 4 + kc) * 512, 32, lane);
        v16h bm = load_b_tile(Bbase + kc * 32, 128, lane);
        acc = wmma16(a, bm, acc);
    }
    int col = ct * 16 + (lane & 15);
    int hi  = lane >> 4;
    for (int r = 0; r < 8; ++r) {
        int row = mt * 16 + r + hi * 8;
        catT2[(size_t)(b * NPIX + col) * 192 + row] = (half_t)(acc[r] + zb[row]);
    }
}

// comb = m_w(192x192) @ concat(zz, hf); output f32 channel-major
__global__ __launch_bounds__(32)
void k_comb(const half_t* __restrict__ Wp, const half_t* __restrict__ catT2,
            const float* __restrict__ mb, float* __restrict__ comb) {
    int lane = threadIdx.x;
    int mt = blockIdx.x, ct = blockIdx.y, b = blockIdx.z;
    const half_t* Bbase = catT2 + (size_t)(b * NPIX + ct * 16) * 192;
    v8f acc = {};
    for (int kc = 0; kc < 6; ++kc) {
        v16h a  = load_a_tile(Wp + (size_t)(mt * 6 + kc) * 512, 32, lane);
        v16h bm = load_b_tile(Bbase + kc * 32, 192, lane);
        acc = wmma16(a, bm, acc);
    }
    int col = ct * 16 + (lane & 15);
    int hi  = lane >> 4;
    for (int r = 0; r < 8; ++r) {
        int row = mt * 16 + r + hi * 8;
        comb[(size_t)(b * 192 + row) * NPIX + col] = acc[r] + mb[row];
    }
}

// Final SAM gates: update carried m (cbuf), next-conv h (Hf16), write d_out
__global__ void k_samfinal(const float* __restrict__ comb, const float* __restrict__ cnew,
                           float* __restrict__ cbuf, half_t* __restrict__ Hf16,
                           float* __restrict__ out, int t) {
    int idx = blockIdx.x * blockDim.x + threadIdx.x;
    if (idx >= BATCH * 64 * NPIX) return;
    int p  = idx & 1023;
    int ch = (idx >> 10) & 63;
    int b  = idx >> 16;
    float mo = comb[(size_t)(b * 192 + ch) * NPIX + p];
    float mg = comb[(size_t)(b * 192 + 64 + ch) * NPIX + p];
    float mi = comb[(size_t)(b * 192 + 128 + ch) * NPIX + p];
    float mf = cnew[idx];
    float si = sigm(mi);
    float m2 = (1.f - si) * mf + si * tanhf(mg);
    float h2 = sigm(mo) * m2;
    cbuf[idx] = m2;
    Hf16[idx] = (half_t)h2;
    out[((size_t)(b * 64 + ch) * SEQ + t) * NPIX + p] = h2;
}

// ---------------------------------------------------------------------------
extern "C" void kernel_launch(void* const* d_in, const int* in_sizes, int n_in,
                              void* d_out, int out_size, void* d_ws, size_t ws_size,
                              hipStream_t stream) {
    const float* X      = (const float*)d_in[0];
    const float* conv_w = (const float*)d_in[1];
    const float* conv_b = (const float*)d_in[2];
    const float* W_ci   = (const float*)d_in[3];
    const float* W_cf   = (const float*)d_in[4];
    const float* W_co   = (const float*)d_in[5];
    const float* q_w    = (const float*)d_in[6];
    const float* q_b    = (const float*)d_in[7];
    const float* k_w    = (const float*)d_in[8];
    const float* k_b    = (const float*)d_in[9];
    const float* k2_w   = (const float*)d_in[10];
    const float* k2_b   = (const float*)d_in[11];
    const float* v_w    = (const float*)d_in[12];
    const float* v_b    = (const float*)d_in[13];
    const float* v2_w   = (const float*)d_in[14];
    const float* v2_b   = (const float*)d_in[15];
    const float* z_w    = (const float*)d_in[16];
    const float* z_b    = (const float*)d_in[17];
    const float* m_w    = (const float*)d_in[18];
    const float* m_b    = (const float*)d_in[19];
    float* out = (float*)d_out;

    char* w = (char*)d_ws;
    auto alloc = [&](size_t bytes) -> char* {
        char* p = w;
        w += (bytes + 255) & ~(size_t)255;
        return p;
    };
    half_t* Xf16  = (half_t*)alloc((size_t)BATCH * 64 * SEQ * NPIX * 2);
    half_t* WcP   = (half_t*)alloc((size_t)16 * 9 * 4 * 512 * 2);
    half_t* WhP   = (half_t*)alloc((size_t)6 * 2 * 512 * 2);
    half_t* WmP   = (half_t*)alloc((size_t)5 * 2 * 512 * 2);
    half_t* WzP   = (half_t*)alloc((size_t)8 * 4 * 512 * 2);
    half_t* WmwP  = (half_t*)alloc((size_t)12 * 6 * 512 * 2);
    half_t* Hf16  = (half_t*)alloc((size_t)BATCH * 64 * NPIX * 2);
    half_t* hT    = (half_t*)alloc((size_t)BATCH * NPIX * 64 * 2);
    half_t* mT    = (half_t*)alloc((size_t)BATCH * NPIX * 64 * 2);
    half_t* catT2 = (half_t*)alloc((size_t)BATCH * NPIX * 192 * 2);
    half_t* zcatT = (half_t*)alloc((size_t)BATCH * NPIX * 128 * 2);
    half_t* qA    = (half_t*)alloc((size_t)BATCH * NPIX * 32 * 2);
    half_t* kT    = (half_t*)alloc((size_t)BATCH * NPIX * 32 * 2);
    half_t* kmT   = (half_t*)alloc((size_t)BATCH * NPIX * 32 * 2);
    half_t* vh    = (half_t*)alloc((size_t)BATCH * 64 * NPIX * 2);
    half_t* vm    = (half_t*)alloc((size_t)BATCH * 64 * NPIX * 2);
    float* logits = (float*)alloc((size_t)BATCH * NPIX * NPIX * 4);
    half_t* attnH = (half_t*)alloc((size_t)BATCH * NPIX * NPIX * 2);
    half_t* attnM = (half_t*)alloc((size_t)BATCH * NPIX * NPIX * 2);
    float* gbuf   = (float*)alloc((size_t)256 * NCOL * 4);
    float* cbuf   = (float*)alloc((size_t)BATCH * 64 * NPIX * 4);
    float* cnew   = (float*)alloc((size_t)BATCH * 64 * NPIX * 4);
    float* combb  = (float*)alloc((size_t)BATCH * 192 * NPIX * 4);

    auto cdiv = [](int a, int b) { return (a + b - 1) / b; };

    // ---- per-call constant preparation (deterministic) ----
    int n = BATCH * 64 * SEQ * NPIX;
    k_f32_to_f16<<<cdiv(n, 256), 256, 0, stream>>>(X, Xf16, n);
    k_pack_conv<<<cdiv(256 * 128 * 9, 256), 256, 0, stream>>>(conv_w, WcP);
    k_pack_a<<<cdiv(16 * 64, 256), 256, 0, stream>>>(q_w, WhP, 16, 64, 0);
    k_pack_a<<<cdiv(16 * 64, 256), 256, 0, stream>>>(k_w, WhP, 16, 64, 1);
    k_pack_a<<<cdiv(64 * 64, 256), 256, 0, stream>>>(v_w, WhP, 64, 64, 2);
    k_pack_a<<<cdiv(16 * 64, 256), 256, 0, stream>>>(k2_w, WmP, 16, 64, 0);
    k_pack_a<<<cdiv(64 * 64, 256), 256, 0, stream>>>(v2_w, WmP, 64, 64, 1);
    k_pack_a<<<cdiv(128 * 128, 256), 256, 0, stream>>>(z_w, WzP, 128, 128, 0);
    k_pack_a<<<cdiv(192 * 192, 256), 256, 0, stream>>>(m_w, WmwP, 192, 192, 0);

    // zero-init: K-pad columns of qA/kT/kmT, initial h (Hf16), initial c (cbuf)
    k_zero_h<<<cdiv(BATCH * NPIX * 32, 256), 256, 0, stream>>>(qA, BATCH * NPIX * 32);
    k_zero_h<<<cdiv(BATCH * NPIX * 32, 256), 256, 0, stream>>>(kT, BATCH * NPIX * 32);
    k_zero_h<<<cdiv(BATCH * NPIX * 32, 256), 256, 0, stream>>>(kmT, BATCH * NPIX * 32);
    k_zero_h<<<cdiv(BATCH * 64 * NPIX, 256), 256, 0, stream>>>(Hf16, BATCH * 64 * NPIX);
    k_zero_f<<<cdiv(BATCH * 64 * NPIX, 256), 256, 0, stream>>>(cbuf, BATCH * 64 * NPIX);

    const int nelem = BATCH * 64 * NPIX;
    for (int t = 0; t < SEQ; ++t) {
        k_conv<<<NCOL / 16, 128, 0, stream>>>(Xf16, Hf16, WcP, conv_b, gbuf, t);
        k_lstm<<<cdiv(nelem, 256), 256, 0, stream>>>(gbuf, cbuf, W_ci, W_cf, W_co,
                                                     cnew, hT, mT, catT2);
        k_projH<<<dim3(6, 64, BATCH), 32, 0, stream>>>(WhP, hT, q_b, k_b, v_b, qA, kT, vh);
        k_projM<<<dim3(5, 64, BATCH), 32, 0, stream>>>(WmP, mT, k2_b, v2_b, kmT, vm);
        k_attn<<<dim3(64, 64, BATCH), 32, 0, stream>>>(qA, kT, logits);
        k_softmax<<<BATCH * NPIX, 256, 0, stream>>>(logits, attnH);
        k_attn<<<dim3(64, 64, BATCH), 32, 0, stream>>>(qA, kmT, logits);
        k_softmax<<<BATCH * NPIX, 256, 0, stream>>>(logits, attnM);
        k_zgemm<<<dim3(64, 4, BATCH), 32, 0, stream>>>(attnH, vh, zcatT, 0);
        k_zgemm<<<dim3(64, 4, BATCH), 32, 0, stream>>>(attnM, vm, zcatT, 64);
        k_zz<<<dim3(8, 64, BATCH), 32, 0, stream>>>(WzP, zcatT, z_b, catT2);
        k_comb<<<dim3(12, 64, BATCH), 32, 0, stream>>>(WmwP, catT2, m_b, combb);
        k_samfinal<<<cdiv(nelem, 256), 256, 0, stream>>>(combb, cnew, cbuf, Hf16, out, t);
    }
    (void)in_sizes; (void)n_in; (void)out_size; (void)ws_size;
}
